// Transformer_63230508531941
// MI455X (gfx1250) — compile-verified, matrix operand order checked by reading, not run on hardware
//
#include <hip/hip_runtime.h>
#include <hip/hip_bf16.h>

typedef __attribute__((ext_vector_type(16))) _Float16 v16h;
typedef __attribute__((ext_vector_type(8)))  _Float16 v8h;
typedef __attribute__((ext_vector_type(8)))  float    v8f;
typedef __attribute__((ext_vector_type(4)))  int      v4i;

#define D_MODEL 512
#define N_HEAD  8
#define D_K     64
#define SEQ     96
#define NLAYERS 6
#define LN_EPS  1e-5f

// ---------------------------------------------------------------------------
// Async global->LDS staging (gfx1250 ASYNCcnt path) when the toolchain has it.
// Builtin signature (probe-confirmed): (AS1 v4i*, AS3 v4i*, imm offset, imm cpol)
// ---------------------------------------------------------------------------
#if defined(__gfx1250__) && __has_builtin(__builtin_amdgcn_global_load_async_to_lds_b128)
#define HAS_ASYNC_LDS 1
typedef __attribute__((address_space(1))) v4i as1_v4i;
typedef __attribute__((address_space(3))) v4i as3_v4i;
__device__ __forceinline__ void wait_async_zero() {
#if __has_builtin(__builtin_amdgcn_s_wait_asynccnt)
    __builtin_amdgcn_s_wait_asynccnt(0);
#else
    asm volatile("s_wait_asynccnt 0" ::: "memory");
#endif
}
#else
#define HAS_ASYNC_LDS 0
#endif

// ---------------------------------------------------------------------------
// Weight pre-pass: W [K,N] fp32 (row-major)  ->  Wt [N,K] f16 (row-major).
// Makes GEMM B-staging a straight aligned copy (no transpose, no convert).
// ---------------------------------------------------------------------------
__global__ void transpose_to_f16(const float* __restrict__ W, _Float16* __restrict__ Wt,
                                 int K, int N)
{
    int idx = blockIdx.x * blockDim.x + threadIdx.x;
    if (idx >= K * N) return;
    int n = idx / K;             // output row (coalesced writes along k)
    int k = idx - n * K;
    Wt[(size_t)n * K + k] = (_Float16)W[(size_t)k * N + n];
}

// ---------------------------------------------------------------------------
// GEMM, f16 WMMA compute:  C = act(A @ W + bias)
//   A  : [M,K] fp32            Wt : [N,K] f16 (pre-transposed)
// Block tile 128(M) x 128(N), K step 32. 256 threads = 8 waves;
// wave = 32(M) x 64(N) patch -> 2x4 WMMA tiles, 8 v_wmma per K-step.
// M, N, K must be multiples of 128 / 128 / 32 (true for all call sites).
// flags: bit0 = add bias, bit1 = relu
// ---------------------------------------------------------------------------
__global__ __launch_bounds__(256)
void gemm_f16wmma(const float* __restrict__ A, const _Float16* __restrict__ Wt,
                  const float* __restrict__ bias, float* __restrict__ C,
                  int M, int K, int N, int flags)
{
    __shared__ _Float16 As[128][40];   // 32 K-halves + 8 pad (stride 80B, 16B mult)
    __shared__ _Float16 Bs[128][40];

    const int tid  = threadIdx.x;
    const int wave = tid >> 5;
    const int lane = tid & 31;
    const int l15  = lane & 15;
    const int hi   = lane >> 4;        // 0 or 1
    const int bm   = blockIdx.x * 128;
    const int bn   = blockIdx.y * 128;
    const int wm   = (wave & 3) * 32;  // 4 waves along M
    const int wn   = (wave >> 2) * 64; // 2 waves along N

    v8f acc[2][4];
#pragma unroll
    for (int i = 0; i < 2; ++i)
#pragma unroll
        for (int j = 0; j < 4; ++j) acc[i][j] = (v8f){};

    for (int k0 = 0; k0 < K; k0 += 32) {
        // --- Stage B tile: 128 rows(n) x 32 halves(k), 16B-aligned chunks ---
#pragma unroll
        for (int j = 0; j < 2; ++j) {
            int q  = tid * 2 + j;      // 0..511 -> 512 x b128
            int rr = q >> 2;           // 0..127
            int ss = (q & 3) * 8;      // half offset 0/8/16/24
            const _Float16* gp = Wt + (size_t)(bn + rr) * K + k0 + ss;
#if HAS_ASYNC_LDS
            __builtin_amdgcn_global_load_async_to_lds_b128(
                (as1_v4i*)gp, (as3_v4i*)&Bs[rr][ss], 0, 0);
#else
            *(v8h*)&Bs[rr][ss] = *(const v8h*)gp;
#endif
        }
        // --- Stage A tile: 128 rows x 32 halves; thread -> 16 contiguous f32 ---
        {
            int r = tid >> 1;
            int c = (tid & 1) * 16;
            const float* src = A + (size_t)(bm + r) * K + k0 + c;
#pragma unroll
            for (int i = 0; i < 16; ++i) As[r][c + i] = (_Float16)src[i];
        }
#if HAS_ASYNC_LDS
        wait_async_zero();
#endif
        __syncthreads();

        // Fragments (lane<16: row m, K{0..7,16..23}; lane>=16: K{8..15,24..31})
        v16h af[2], bf[4];
#pragma unroll
        for (int i = 0; i < 2; ++i) {
            const _Float16* pa = &As[wm + i * 16 + l15][hi * 8];
            v8h lo = *(const v8h*)pa;
            v8h hh = *(const v8h*)(pa + 16);
            v16h f;
#pragma unroll
            for (int e = 0; e < 8; ++e) { f[e] = lo[e]; f[8 + e] = hh[e]; }
            af[i] = f;
        }
#pragma unroll
        for (int j = 0; j < 4; ++j) {
            const _Float16* pb = &Bs[wn + j * 16 + l15][hi * 8];
            v8h lo = *(const v8h*)pb;
            v8h hh = *(const v8h*)(pb + 16);
            v16h f;
#pragma unroll
            for (int e = 0; e < 8; ++e) { f[e] = lo[e]; f[8 + e] = hh[e]; }
            bf[j] = f;
        }
#pragma unroll
        for (int i = 0; i < 2; ++i)
#pragma unroll
            for (int j = 0; j < 4; ++j)
                acc[i][j] = __builtin_amdgcn_wmma_f32_16x16x32_f16(
                    false, af[i], false, bf[j], (short)0, acc[i][j], false, false);
        __syncthreads();
    }

    // Epilogue. C layout: elem r, lanes 0-15 -> row r, lanes 16-31 -> row r+8.
#pragma unroll
    for (int i = 0; i < 2; ++i) {
#pragma unroll
        for (int j = 0; j < 4; ++j) {
            int col  = bn + wn + j * 16 + l15;
            int row0 = bm + wm + i * 16 + hi * 8;
            float bv = (flags & 1) ? bias[col] : 0.0f;
#pragma unroll
            for (int r = 0; r < 8; ++r) {
                float v = acc[i][j][r] + bv;
                if (flags & 2) v = fmaxf(v, 0.0f);
                C[(size_t)(row0 + r) * N + col] = v;
            }
        }
    }
}

// ---------------------------------------------------------------------------
// Fused attention for one (batch, head): softmax((Q Kt)/8 + mask) V
// q/k/v buffers are [B, L, H*DK] fp32. Block = 192 threads = 6 waves,
// wave w owns q rows [16w, 16w+16).
// ---------------------------------------------------------------------------
__global__ __launch_bounds__(192)
void attn_wmma(const float* __restrict__ qb, const float* __restrict__ kb,
               const float* __restrict__ vb, const int* __restrict__ keytok,
               float* __restrict__ ctx, int causal)
{
    __shared__ _Float16 Ks[SEQ][D_K + 8];      // [n][d]   96 x 72
    __shared__ _Float16 Vt[D_K][SEQ + 8];      // transposed V: 64 x 104
    __shared__ _Float16 Ps[6][16][SEQ + 8];    // per-wave probs tile 16 x 104

    const int b    = blockIdx.x / N_HEAD;
    const int h    = blockIdx.x % N_HEAD;
    const int tid  = threadIdx.x;
    const int wave = tid >> 5;
    const int lane = tid & 31;
    const int l15  = lane & 15;
    const int hi   = lane >> 4;
    const size_t base = ((size_t)b * SEQ) * D_MODEL + h * D_K;

    for (int idx = tid; idx < SEQ * D_K; idx += 192) {
        int r = idx >> 6;
        int c = idx & 63;
        Ks[r][c] = (_Float16)kb[base + (size_t)r * D_MODEL + c];
        Vt[c][r] = (_Float16)vb[base + (size_t)r * D_MODEL + c];
    }
    __syncthreads();

    const int qrow0 = wave * 16;

    v16h qf[2];
#pragma unroll
    for (int ks = 0; ks < 2; ++ks) {
        const float* src = qb + base + (size_t)(qrow0 + l15) * D_MODEL + ks * 32 + hi * 8;
        v16h f;
#pragma unroll
        for (int e = 0; e < 8; ++e) { f[e] = (_Float16)src[e]; f[8 + e] = (_Float16)src[16 + e]; }
        qf[ks] = f;
    }

    v8f sc[6];
#pragma unroll
    for (int nt = 0; nt < 6; ++nt) sc[nt] = (v8f){};
#pragma unroll
    for (int ks = 0; ks < 2; ++ks) {
#pragma unroll
        for (int nt = 0; nt < 6; ++nt) {
            const _Float16* pk = &Ks[nt * 16 + l15][ks * 32 + hi * 8];
            v8h lo = *(const v8h*)pk;
            v8h hh = *(const v8h*)(pk + 16);
            v16h kf;
#pragma unroll
            for (int e = 0; e < 8; ++e) { kf[e] = lo[e]; kf[8 + e] = hh[e]; }
            sc[nt] = __builtin_amdgcn_wmma_f32_16x16x32_f16(
                false, qf[ks], false, kf, (short)0, sc[nt], false, false);
        }
    }

    int km[6];
#pragma unroll
    for (int nt = 0; nt < 6; ++nt) km[nt] = (keytok[b * SEQ + nt * 16 + l15] == 0);

    const float scale = 0.125f;  // 1/sqrt(64)
    float mx[8], sm[8];
#pragma unroll
    for (int r = 0; r < 8; ++r) mx[r] = -1e30f;
#pragma unroll
    for (int nt = 0; nt < 6; ++nt) {
        int col = nt * 16 + l15;
#pragma unroll
        for (int r = 0; r < 8; ++r) {
            int row = qrow0 + hi * 8 + r;
            float v = sc[nt][r] * scale;
            if (km[nt] || (causal && col > row)) v = -1e30f;
            sc[nt][r] = v;
            mx[r] = fmaxf(mx[r], v);
        }
    }
#pragma unroll
    for (int s = 1; s < 16; s <<= 1)
#pragma unroll
        for (int r = 0; r < 8; ++r) mx[r] = fmaxf(mx[r], __shfl_xor(mx[r], s));
#pragma unroll
    for (int r = 0; r < 8; ++r) sm[r] = 0.0f;
#pragma unroll
    for (int nt = 0; nt < 6; ++nt)
#pragma unroll
        for (int r = 0; r < 8; ++r) {
            float e = __expf(sc[nt][r] - mx[r]);
            sc[nt][r] = e;
            sm[r] += e;
        }
#pragma unroll
    for (int s = 1; s < 16; s <<= 1)
#pragma unroll
        for (int r = 0; r < 8; ++r) sm[r] += __shfl_xor(sm[r], s);
    float inv[8];
#pragma unroll
    for (int r = 0; r < 8; ++r) inv[r] = 1.0f / sm[r];

#pragma unroll
    for (int nt = 0; nt < 6; ++nt)
#pragma unroll
        for (int r = 0; r < 8; ++r)
            Ps[wave][hi * 8 + r][nt * 16 + l15] = (_Float16)(sc[nt][r] * inv[r]);
    __syncthreads();

    v8f oc[4];
#pragma unroll
    for (int nt = 0; nt < 4; ++nt) oc[nt] = (v8f){};
#pragma unroll
    for (int ks = 0; ks < 3; ++ks) {
        const _Float16* pp = &Ps[wave][l15][ks * 32 + hi * 8];
        v8h plo = *(const v8h*)pp;
        v8h phh = *(const v8h*)(pp + 16);
        v16h pf;
#pragma unroll
        for (int e = 0; e < 8; ++e) { pf[e] = plo[e]; pf[8 + e] = phh[e]; }
#pragma unroll
        for (int nt = 0; nt < 4; ++nt) {
            const _Float16* pv = &Vt[nt * 16 + l15][ks * 32 + hi * 8];
            v8h lo = *(const v8h*)pv;
            v8h hh = *(const v8h*)(pv + 16);
            v16h vf;
#pragma unroll
            for (int e = 0; e < 8; ++e) { vf[e] = lo[e]; vf[8 + e] = hh[e]; }
            oc[nt] = __builtin_amdgcn_wmma_f32_16x16x32_f16(
                false, pf, false, vf, (short)0, oc[nt], false, false);
        }
    }

#pragma unroll
    for (int nt = 0; nt < 4; ++nt)
#pragma unroll
        for (int r = 0; r < 8; ++r) {
            int row = qrow0 + hi * 8 + r;
            ctx[((size_t)b * SEQ + row) * D_MODEL + h * D_K + nt * 16 + l15] = oc[nt][r];
        }
}

// ---------------------------------------------------------------------------
// out = LayerNorm(x + res) * g + b    (one block per row of 512)
// ---------------------------------------------------------------------------
__global__ __launch_bounds__(256)
void add_ln(const float* __restrict__ x, const float* __restrict__ res,
            const float* __restrict__ g, const float* __restrict__ beta,
            float* __restrict__ out)
{
    __shared__ float rs[8], rq[8];
    __shared__ float mu_s, rstd_s;
    const int row = blockIdx.x;
    const int tid = threadIdx.x;
    const size_t off = (size_t)row * D_MODEL;

    float v0 = x[off + tid]       + res[off + tid];
    float v1 = x[off + tid + 256] + res[off + tid + 256];
    float s  = v0 + v1;
    float sq = v0 * v0 + v1 * v1;
#pragma unroll
    for (int m = 1; m < 32; m <<= 1) { s += __shfl_xor(s, m); sq += __shfl_xor(sq, m); }
    if ((tid & 31) == 0) { rs[tid >> 5] = s; rq[tid >> 5] = sq; }
    __syncthreads();
    if (tid == 0) {
        float S = 0.0f, Q = 0.0f;
#pragma unroll
        for (int i = 0; i < 8; ++i) { S += rs[i]; Q += rq[i]; }
        float mu  = S * (1.0f / D_MODEL);
        float var = Q * (1.0f / D_MODEL) - mu * mu;
        mu_s   = mu;
        rstd_s = rsqrtf(var + LN_EPS);
    }
    __syncthreads();
    out[off + tid]       = (v0 - mu_s) * rstd_s * g[tid]       + beta[tid];
    out[off + tid + 256] = (v1 - mu_s) * rstd_s * g[tid + 256] + beta[tid + 256];
}

// ---------------------------------------------------------------------------
// Embedding gather + sinusoidal positional encoding
// ---------------------------------------------------------------------------
__global__ void embed_pe(const int* __restrict__ tok, const float* __restrict__ emb,
                         float* __restrict__ out, int rows, int Lseq)
{
    int idx = blockIdx.x * blockDim.x + threadIdx.x;
    if (idx >= rows * D_MODEL) return;
    int m = idx >> 9;
    int d = idx & (D_MODEL - 1);
    int l = m % Lseq;
    int t = tok[m];
    float freq = __expf(-(float)(d & ~1) * (9.210340371976184f / (float)D_MODEL));
    float ang  = (float)l * freq;
    float pe   = (d & 1) ? __cosf(ang) : __sinf(ang);
    out[idx] = emb[(size_t)t * D_MODEL + d] + pe;
}

// ---------------------------------------------------------------------------
// Final head: out[m, j] = y[m,:] . head_w[:, j] + head_b[j]   (N=9)
// ---------------------------------------------------------------------------
__global__ void head_proj(const float* __restrict__ y, const float* __restrict__ hw,
                          const float* __restrict__ hb, float* __restrict__ out, int M)
{
    int idx = blockIdx.x * blockDim.x + threadIdx.x;
    if (idx >= M * 9) return;
    int m = idx / 9, j = idx % 9;
    const float* py = y + (size_t)m * D_MODEL;
    float acc = hb[j];
    for (int d = 0; d < D_MODEL; ++d) acc += py[d] * hw[d * 9 + j];
    out[idx] = acc;
}

// ---------------------------------------------------------------------------
// Host orchestration
// ---------------------------------------------------------------------------
struct MhaP { const float *ln_b, *ln_g, *wk, *wo, *wq, *wv; };
struct FfnP { const float *b1, *b2, *ln_b, *ln_g, *w1, *w2; };

extern "C" void kernel_launch(void* const* d_in, const int* in_sizes, int n_in,
                              void* d_out, int out_size, void* d_ws, size_t ws_size,
                              hipStream_t stream) {
    (void)in_sizes; (void)n_in; (void)out_size; (void)ws_size;
    const int B = 128, L = SEQ, M = B * L;          // 12288 rows

    const int* enc_tok = (const int*)d_in[0];
    const int* dec_tok = (const int*)d_in[1];

    // params flattened as jax pytree: dict keys alphabetical, lists in order.
    auto F = [&](int i) { return (const float*)d_in[i]; };
    int i = 2;
    MhaP dec_self[NLAYERS], dec_cross[NLAYERS], enc_mha[NLAYERS];
    FfnP dec_ffn[NLAYERS], enc_ffn[NLAYERS];
    for (int l = 0; l < NLAYERS; ++l) {             // 'dec': cross, ffn, self
        dec_cross[l] = { F(i), F(i+1), F(i+2), F(i+3), F(i+4), F(i+5) }; i += 6;
        dec_ffn[l]   = { F(i), F(i+1), F(i+2), F(i+3), F(i+4), F(i+5) }; i += 6;
        dec_self[l]  = { F(i), F(i+1), F(i+2), F(i+3), F(i+4), F(i+5) }; i += 6;
    }
    const float* dec_emb = F(i++);                  // 'dec_emb'
    for (int l = 0; l < NLAYERS; ++l) {             // 'enc': ffn, mha
        enc_ffn[l] = { F(i), F(i+1), F(i+2), F(i+3), F(i+4), F(i+5) }; i += 6;
        enc_mha[l] = { F(i), F(i+1), F(i+2), F(i+3), F(i+4), F(i+5) }; i += 6;
    }
    const float* enc_emb = F(i++);                  // 'enc_emb'
    const float* head_b  = F(i++);                  // 'head_b'
    const float* head_w  = F(i++);                  // 'head_w'

    // workspace carve
    const size_t S = (size_t)M * D_MODEL;
    float* x    = (float*)d_ws;      // encoder stream
    float* y    = x + S;             // decoder stream
    float* qb   = y + S;
    float* kb   = qb + S;
    float* vb   = kb + S;
    float* ctx  = vb + S;
    float* t1   = ctx + S;
    float* hid  = t1 + S;            // M x 2048
    _Float16* wt = (_Float16*)(hid + (size_t)M * 2048);  // 2048*512 halves scratch

    auto gemm = [&](const float* A, const float* W, const float* bias, float* C,
                    int Mm, int K, int N, int flags) {
        transpose_to_f16<<<(K * N + 255) / 256, 256, 0, stream>>>(W, wt, K, N);
        dim3 g(Mm / 128, N / 128);
        gemm_f16wmma<<<g, 256, 0, stream>>>(A, wt, bias, C, Mm, K, N, flags);
    };
    auto mha = [&](float* io, const float* kvsrc, const int* keytok, int causal,
                   const MhaP& p) {
        gemm(io,    p.wq, nullptr, qb, M, D_MODEL, D_MODEL, 0);
        gemm(kvsrc, p.wk, nullptr, kb, M, D_MODEL, D_MODEL, 0);
        gemm(kvsrc, p.wv, nullptr, vb, M, D_MODEL, D_MODEL, 0);
        attn_wmma<<<B * N_HEAD, 192, 0, stream>>>(qb, kb, vb, keytok, ctx, causal);
        gemm(ctx, p.wo, nullptr, t1, M, D_MODEL, D_MODEL, 0);
        add_ln<<<M, 256, 0, stream>>>(t1, io, p.ln_g, p.ln_b, io);
    };
    auto ffn = [&](float* io, const FfnP& p) {
        gemm(io,  p.w1, p.b1, hid, M, D_MODEL, 4 * D_MODEL, 3);  // bias + relu
        gemm(hid, p.w2, p.b2, t1,  M, 4 * D_MODEL, D_MODEL, 1);  // bias
        add_ln<<<M, 256, 0, stream>>>(t1, io, p.ln_g, p.ln_b, io);
    };

    const int embThreads = 256;
    embed_pe<<<(M * D_MODEL + embThreads - 1) / embThreads, embThreads, 0, stream>>>(
        enc_tok, enc_emb, x, M, L);
    for (int l = 0; l < NLAYERS; ++l) {
        mha(x, x, enc_tok, 0, enc_mha[l]);
        ffn(x, enc_ffn[l]);
    }

    embed_pe<<<(M * D_MODEL + embThreads - 1) / embThreads, embThreads, 0, stream>>>(
        dec_tok, dec_emb, y, M, L);
    for (int l = 0; l < NLAYERS; ++l) {
        mha(y, y, dec_tok, 1, dec_self[l]);   // causal self-attn
        mha(y, x, enc_tok, 0, dec_cross[l]);  // cross-attn over encoder output
        ffn(y, dec_ffn[l]);
    }

    head_proj<<<(M * 9 + 255) / 256, 256, 0, stream>>>(y, head_w, head_b,
                                                       (float*)d_out, M);
}
